// EfficientAttention_26697516711947
// MI455X (gfx1250) — matrix operand order repaired and checked by reading, not compiled, not run on hardware
//
#include <hip/hip_runtime.h>
#include <cstdint>

// ---------------------------------------------------------------------------
// MI455X (gfx1250). bf16 WMMA pipeline, fp32 accumulate:
//   1. cast x, w_qkv, w_proj -> bf16
//   2. x11 = softmax(gn_b)         (exact collapse of the coord-attn branch:
//      spatial mean of a group-normalized tensor is 0 -> mean = gn_b)
//   3. qkv  = Wqkv @ x   (768x256x4096 per b)  WMMA, double-buffered LDS,
//      A-tile staged via GLOBAL_LOAD_ASYNC_TO_LDS_B128 (ASYNCcnt)
//   4. q,k row reciprocal L2 norms
//   5. attn = softmax(q^ k^T * temp) per (b,h)  WMMA, K=4096
//   6. outa = attn @ v                          WMMA
//   7. out  = Wproj @ outa -> f32 d_out         WMMA, same async GEMM
//   8. fuse: out = x * sigmoid(sum_c x11[c]*out[g,c]) + out
// ~36.5 GFLOP, ~700MB total traffic, L2-resident (192MB): memory-bound at
// 23.3 TB/s once GEMMs ride the bf16 WMMA pipe. Workspace ~169 MB.
// ---------------------------------------------------------------------------

typedef unsigned short u16;
typedef unsigned int   u32;
typedef __attribute__((ext_vector_type(16))) __bf16 v16bf;
typedef __attribute__((ext_vector_type(8)))  float  v8f;
typedef __attribute__((ext_vector_type(4)))  int    v4i;

union Frag { v16bf v; u32 u[8]; };

__device__ __forceinline__ u16 f2bf(float f) {
  union { float f; u32 u; } c; c.f = f;
  u32 r = c.u + 0x7fffu + ((c.u >> 16) & 1u);   // round-to-nearest-even
  return (u16)(r >> 16);
}
__device__ __forceinline__ float bf2f(u16 h) {
  union { u32 u; float f; } c; c.u = ((u32)h) << 16;
  return c.f;
}

// ---- CDNA5 async global->LDS copy (16B per lane), ASYNCcnt tracked -------
__device__ __forceinline__ void async_copy_b128(const void* gsrc, void* ldst) {
#if defined(__has_builtin) && __has_builtin(__builtin_amdgcn_global_load_async_to_lds_b128)
  __builtin_amdgcn_global_load_async_to_lds_b128(
      (__attribute__((address_space(1))) v4i*)(uintptr_t)gsrc,
      (__attribute__((address_space(3))) v4i*)(u32)(uintptr_t)ldst,
      0, 0);
#else
  asm volatile("global_load_async_to_lds_b128 %0, %1, off"
               :: "v"((u32)(uintptr_t)ldst), "v"(gsrc) : "memory");
#endif
}
__device__ __forceinline__ void wait_async0() {
#if defined(__has_builtin) && __has_builtin(__builtin_amdgcn_s_wait_asynccnt)
  __builtin_amdgcn_s_wait_asynccnt(0);
#else
  asm volatile("s_wait_asynccnt 0" ::: "memory");
#endif
}

// ---------------- casts ----------------
__global__ __launch_bounds__(256) void cast_bf16_kernel(const float* __restrict__ src,
                                                        u16* __restrict__ dst, int n) {
  int i = (blockIdx.x * 256 + threadIdx.x) * 4;
  if (i + 3 < n) {
    float4 f = *(const float4*)(src + i);
    u32 lo = (u32)f2bf(f.x) | ((u32)f2bf(f.y) << 16);
    u32 hi = (u32)f2bf(f.z) | ((u32)f2bf(f.w) << 16);
    *(uint2*)(dst + i) = make_uint2(lo, hi);
  }
}

// x11 = softmax(gn_b)
__global__ void x11_kernel(const float* __restrict__ gnb, float* __restrict__ x11) {
  if (threadIdx.x == 0) {
    float v[16], mx = -1e30f;
    for (int c = 0; c < 16; ++c) { v[c] = gnb[c]; mx = fmaxf(mx, v[c]); }
    float s = 0.f;
    for (int c = 0; c < 16; ++c) { v[c] = __expf(v[c] - mx); s += v[c]; }
    float inv = 1.f / s;
    for (int c = 0; c < 16; ++c) x11[c] = v[c] * inv;
  }
}

// ---------------- batched GEMM: O[b][m][n] = sum_k W[m][k] * X[b][k][n] ----
// K=256. 128x128 tile / 256-thread block. Double-buffered LDS; A staged by
// async-to-LDS DMA, B staged via VGPRs (transpose+pack). bf16 WMMA 16x16x32.
#define LDSA_S 72
#define LDSB_S 72

template <bool OUT_F32>
__global__ __launch_bounds__(256) void gemm_kn(const u16* __restrict__ Wmat,  // [M][256]
                                               const u16* __restrict__ X,     // [B][256][4096]
                                               u16* __restrict__ Obf,
                                               float* __restrict__ Of,
                                               int M) {
  __shared__ u16 ldsA[2][128 * LDSA_S];
  __shared__ u16 ldsB[2][128 * LDSB_S];
  const int tilesM = M >> 7;
  const int bidx = blockIdx.x;
  const int nb = bidx & 31;
  const int mb = (bidx >> 5) % tilesM;
  const int b  = bidx / (32 * tilesM);
  const int o0 = mb * 128, n0 = nb * 128;
  const int tid = threadIdx.x, lane = tid & 31, wid = tid >> 5;
  const int wm = (wid & 3) * 32;   // 4 waves over 128 M
  const int wn = (wid >> 2) * 64;  // 2 waves over 128 N
  const int lh = lane >> 4, r = lane & 15;

  v8f acc[2][4];
  for (int i = 0; i < 2; ++i)
    for (int j = 0; j < 4; ++j)
      for (int e = 0; e < 8; ++e) acc[i][j][e] = 0.f;

  const u16* xb = X + ((size_t)b * 256) * 4096;
  uint4 bregs[4];

  auto issueA = [&](int kc, int buf) {          // 128x64 A tile, async DMA
#pragma unroll
    for (int t = 0; t < 4; ++t) {
      int i = tid + t * 256;
      int row = i >> 3, ch = i & 7;
      async_copy_b128(Wmat + (o0 + row) * 256 + kc + ch * 8,
                      &ldsA[buf][row * LDSA_S + ch * 8]);
    }
  };
  auto loadB = [&](int kc) {                    // 64x128 B tile -> regs
#pragma unroll
    for (int t = 0; t < 4; ++t) {
      int i = tid + t * 256;
      int ki = i >> 4, ch = i & 15;
      bregs[t] = *(const uint4*)(xb + (size_t)(kc + ki) * 4096 + n0 + ch * 8);
    }
  };
  auto storeB = [&](int buf) {                  // regs -> ldsB[n][k]
#pragma unroll
    for (int t = 0; t < 4; ++t) {
      int i = tid + t * 256;
      int ki = i >> 4, ch = i & 15;
      const u16* e = (const u16*)&bregs[t];
#pragma unroll
      for (int q = 0; q < 8; ++q) ldsB[buf][(ch * 8 + q) * LDSB_S + ki] = e[q];
    }
  };
  auto compute = [&](int buf) {
#pragma unroll
    for (int kk = 0; kk < 64; kk += 32) {
      Frag a[2], fb[4];
#pragma unroll
      for (int mi = 0; mi < 2; ++mi) {
        int row = wm + mi * 16 + r;
#pragma unroll
        for (int j = 0; j < 8; ++j) {
          int kb = kk + lh * 16 + 2 * j;
          a[mi].u[j] = *(const u32*)(&ldsA[buf][row * LDSA_S + kb]);
        }
      }
#pragma unroll
      for (int ni = 0; ni < 4; ++ni) {
        int col = wn + ni * 16 + r;
#pragma unroll
        for (int j = 0; j < 8; ++j) {
          int kb = kk + lh * 16 + 2 * j;
          fb[ni].u[j] = *(const u32*)(&ldsB[buf][col * LDSB_S + kb]);
        }
      }
#pragma unroll
      for (int mi = 0; mi < 2; ++mi)
#pragma unroll
        for (int ni = 0; ni < 4; ++ni)
          acc[mi][ni] = __builtin_amdgcn_wmma_f32_16x16x32_bf16(
              false, a[mi].v, false, fb[ni].v, (short)0, acc[mi][ni], false, false);
    }
  };

  // prologue: stage chunk 0
  issueA(0, 0);
  loadB(0);
  storeB(0);
  wait_async0();
  __syncthreads();

  for (int kc = 0; kc < 256; kc += 64) {
    const int buf = (kc >> 6) & 1;
    const int nxt = kc + 64;
    if (nxt < 256) {                 // overlap next-chunk staging with WMMA
      issueA(nxt, buf ^ 1);
      loadB(nxt);
    }
    compute(buf);
    if (nxt < 256) {
      storeB(buf ^ 1);
      wait_async0();
      __syncthreads();
    }
  }

#pragma unroll
  for (int mi = 0; mi < 2; ++mi)
#pragma unroll
    for (int ni = 0; ni < 4; ++ni)
#pragma unroll
      for (int v = 0; v < 8; ++v) {
        int m = o0 + wm + mi * 16 + lh * 8 + v;
        int n = n0 + wn + ni * 16 + r;
        float val = acc[mi][ni][v];
        if (OUT_F32) Of[((size_t)b * M + m) * 4096 + n] = val;
        else         Obf[((size_t)b * M + m) * 4096 + n] = f2bf(val);
      }
}

// ---------------- q/k row reciprocal L2 norms ----------------
__global__ __launch_bounds__(256) void rownorm_kernel(const u16* __restrict__ qkv,
                                                      float* __restrict__ rq,
                                                      float* __restrict__ rk) {
  int row = blockIdx.x * 8 + (threadIdx.x >> 5);   // 0..8191
  int lane = threadIdx.x & 31;
  int b = row >> 9, o = row & 511;                 // q: o<256, k: o>=256
  const u16* p = qkv + ((size_t)b * 768 + o) * 4096;
  float s = 0.f;
  for (int i = lane * 8; i < 4096; i += 256) {
    uint4 vv = *(const uint4*)(p + i);
    const u16* e = (const u16*)&vv;
#pragma unroll
    for (int t = 0; t < 8; ++t) { float f = bf2f(e[t]); s += f * f; }
  }
#pragma unroll
  for (int off = 16; off; off >>= 1) s += __shfl_xor(s, off, 32);
  if (lane == 0) {
    float rv = 1.f / fmaxf(sqrtf(s), 1e-12f);
    if (o < 256) rq[b * 256 + o] = rv;
    else         rk[b * 256 + (o - 256)] = rv;
  }
}

// ---------------- attn = softmax(q^ k^T * temp), 32x32 per (b,h) ----------
__global__ __launch_bounds__(256) void attn_kernel(const u16* __restrict__ qkv,
                                                   const float* __restrict__ rq,
                                                   const float* __restrict__ rk,
                                                   const float* __restrict__ temp,
                                                   u16* __restrict__ attn_out) {
  __shared__ float red[8][32][32];
  int bh = blockIdx.x, b = bh >> 3, h = bh & 7;
  int tid = threadIdx.x, lane = tid & 31, wid = tid >> 5;
  int lh = lane >> 4, r = lane & 15;
  const u16* qbase = qkv + ((size_t)b * 768 + h * 32) * 4096;
  const u16* kbase = qkv + ((size_t)b * 768 + 256 + h * 32) * 4096;

  v8f acc[2][2];
  for (int i = 0; i < 2; ++i)
    for (int j = 0; j < 2; ++j)
      for (int e = 0; e < 8; ++e) acc[i][j][e] = 0.f;

  int k0w = wid * 512;                 // each wave owns 512 of K=4096
  for (int kk = k0w; kk < k0w + 512; kk += 32) {
    Frag a[2], fb[2];
#pragma unroll
    for (int mi = 0; mi < 2; ++mi) {
      int row = mi * 16 + r;
#pragma unroll
      for (int j = 0; j < 8; ++j) {
        int kb = kk + lh * 16 + 2 * j;
        a[mi].u[j] = *(const u32*)(qbase + (size_t)row * 4096 + kb);
      }
    }
#pragma unroll
    for (int ni = 0; ni < 2; ++ni) {
      int row = ni * 16 + r;
#pragma unroll
      for (int j = 0; j < 8; ++j) {
        int kb = kk + lh * 16 + 2 * j;
        fb[ni].u[j] = *(const u32*)(kbase + (size_t)row * 4096 + kb);
      }
    }
#pragma unroll
    for (int mi = 0; mi < 2; ++mi)
#pragma unroll
      for (int ni = 0; ni < 2; ++ni)
        acc[mi][ni] = __builtin_amdgcn_wmma_f32_16x16x32_bf16(
            false, a[mi].v, false, fb[ni].v, (short)0, acc[mi][ni], false, false);
  }
#pragma unroll
  for (int mi = 0; mi < 2; ++mi)
#pragma unroll
    for (int ni = 0; ni < 2; ++ni)
#pragma unroll
      for (int v = 0; v < 8; ++v)
        red[wid][mi * 16 + lh * 8 + v][ni * 16 + r] = acc[mi][ni][v];
  __syncthreads();

  if (tid < 32) {                      // one lane per output row c
    int c = tid;
    float row[32];
    float tval = temp[h];
    float rqv = rq[b * 256 + h * 32 + c];
    float mx = -1e30f;
#pragma unroll
    for (int d = 0; d < 32; ++d) {
      float s = 0.f;
#pragma unroll
      for (int w2 = 0; w2 < 8; ++w2) s += red[w2][c][d];
      s = s * rqv * rk[b * 256 + h * 32 + d] * tval;
      row[d] = s; mx = fmaxf(mx, s);
    }
    float sum = 0.f;
#pragma unroll
    for (int d = 0; d < 32; ++d) { float e = __expf(row[d] - mx); row[d] = e; sum += e; }
    float inv = 1.f / sum;
#pragma unroll
    for (int d = 0; d < 32; ++d)
      attn_out[((size_t)bh * 32 + c) * 32 + d] = f2bf(row[d] * inv);
  }
}

// ---------------- outa = attn @ v, per (b,h), K=32 single WMMA step -------
#define LDSV_S 34
__global__ __launch_bounds__(256) void av_kernel(const u16* __restrict__ qkv,
                                                 const u16* __restrict__ attn,
                                                 u16* __restrict__ outa) {
  __shared__ u16 ldsV[512 * LDSV_S];
  int blk = blockIdx.x;                   // b*64 + h*8 + nb
  int nb = blk & 7, h = (blk >> 3) & 7, b = blk >> 6;
  int n0 = nb * 512;
  const u16* vbase = qkv + ((size_t)b * 768 + 512 + h * 32) * 4096;
  int tid = threadIdx.x, lane = tid & 31, wid = tid >> 5;
  int lh = lane >> 4, r = lane & 15;

  for (int i = tid; i < 32 * 64; i += 256) {     // stage v (32x512) -> [n][d]
    int d = i >> 6, ch = i & 63;
    uint4 vv = *(const uint4*)(vbase + (size_t)d * 4096 + n0 + ch * 8);
    const u16* e = (const u16*)&vv;
#pragma unroll
    for (int t = 0; t < 8; ++t) ldsV[(ch * 8 + t) * LDSV_S + d] = e[t];
  }
  __syncthreads();

  Frag a[2];
  const u16* abase = attn + ((size_t)(b * 8 + h) * 32) * 32;
#pragma unroll
  for (int mi = 0; mi < 2; ++mi) {
    int row = mi * 16 + r;
#pragma unroll
    for (int j = 0; j < 8; ++j) {
      int kb = lh * 16 + 2 * j;
      a[mi].u[j] = *(const u32*)(abase + row * 32 + kb);
    }
  }
  int wn = wid * 64;
  v8f acc[2][4];
  for (int i = 0; i < 2; ++i)
    for (int j = 0; j < 4; ++j)
      for (int e = 0; e < 8; ++e) acc[i][j][e] = 0.f;
  Frag fb[4];
#pragma unroll
  for (int ni = 0; ni < 4; ++ni) {
    int nn = wn + ni * 16 + r;
#pragma unroll
    for (int j = 0; j < 8; ++j) {
      int kb = lh * 16 + 2 * j;
      fb[ni].u[j] = *(const u32*)(ldsV + nn * LDSV_S + kb);
    }
  }
#pragma unroll
  for (int mi = 0; mi < 2; ++mi)
#pragma unroll
    for (int ni = 0; ni < 4; ++ni)
      acc[mi][ni] = __builtin_amdgcn_wmma_f32_16x16x32_bf16(
          false, a[mi].v, false, fb[ni].v, (short)0, acc[mi][ni], false, false);

#pragma unroll
  for (int mi = 0; mi < 2; ++mi)
#pragma unroll
    for (int ni = 0; ni < 4; ++ni)
#pragma unroll
      for (int v = 0; v < 8; ++v) {
        int m = mi * 16 + lh * 8 + v;
        int col = n0 + wn + ni * 16 + r;
        outa[((size_t)b * 256 + h * 32 + m) * 4096 + col] = f2bf(acc[mi][ni][v]);
      }
}

// ---------------- final fuse: out = x * sigmoid(sum_c x11*out) + out ------
__global__ __launch_bounds__(256) void fuse_kernel(const float* __restrict__ x,
                                                   const float* __restrict__ x11,
                                                   float* __restrict__ out) {
  int idx = blockIdx.x * 256 + threadIdx.x;   // 0 .. B*G*N-1
  int n = idx & 4095;
  int bg = idx >> 12;
  int b = bg >> 4, g = bg & 15;
  size_t base = ((size_t)(b * 256 + g * 16)) * 4096 + n;
  float ov[16], wsum = 0.f;
#pragma unroll
  for (int c = 0; c < 16; ++c) {
    ov[c] = out[base + (size_t)c * 4096];
    wsum += x11[c] * ov[c];
  }
  float s = 1.f / (1.f + __expf(-wsum));
#pragma unroll
  for (int c = 0; c < 16; ++c)
    out[base + (size_t)c * 4096] = x[base + (size_t)c * 4096] * s + ov[c];
}

// ---------------------------------------------------------------------------
extern "C" void kernel_launch(void* const* d_in, const int* in_sizes, int n_in,
                              void* d_out, int out_size, void* d_ws, size_t ws_size,
                              hipStream_t stream) {
  (void)in_sizes; (void)n_in; (void)out_size; (void)ws_size;
  const float* x      = (const float*)d_in[0];
  const float* gn_b   = (const float*)d_in[4];
  const float* w_qkv  = (const float*)d_in[5];
  const float* w_proj = (const float*)d_in[6];
  const float* temp   = (const float*)d_in[7];
  float* out = (float*)d_out;
  char* ws = (char*)d_ws;

  u16*   xb      = (u16*)(ws);                    // 33,554,432 B
  u16*   wqkv_b  = (u16*)(ws + 33554432);         //    393,216 B
  u16*   wproj_b = (u16*)(ws + 33947648);         //    131,072 B
  u16*   qkv_b   = (u16*)(ws + 34078720);         // 100,663,296 B
  u16*   outa_b  = (u16*)(ws + 134742016);        // 33,554,432 B
  float* rq      = (float*)(ws + 168296448);      //     16,384 B
  float* rk      = (float*)(ws + 168312832);      //     16,384 B
  u16*   attn_b  = (u16*)(ws + 168329216);        //    262,144 B
  float* x11     = (float*)(ws + 168591360);      //         64 B

  cast_bf16_kernel<<<16384, 256, 0, stream>>>(x,      xb,      16777216);
  cast_bf16_kernel<<<  192, 256, 0, stream>>>(w_qkv,  wqkv_b,  196608);
  cast_bf16_kernel<<<   64, 256, 0, stream>>>(w_proj, wproj_b, 65536);
  x11_kernel<<<1, 32, 0, stream>>>(gn_b, x11);

  gemm_kn<false><<<3072, 256, 0, stream>>>(wqkv_b, xb, qkv_b, nullptr, 768);
  rownorm_kernel<<<1024, 256, 0, stream>>>(qkv_b, rq, rk);
  attn_kernel<<<128, 256, 0, stream>>>(qkv_b, rq, rk, temp, attn_b);
  av_kernel<<<1024, 256, 0, stream>>>(qkv_b, attn_b, outa_b);
  gemm_kn<true><<<1024, 256, 0, stream>>>(wproj_b, outa_b, nullptr, out, 256);
  fuse_kernel<<<4096, 256, 0, stream>>>(x, x11, out);
}
